// RPGNN_33474975105507
// MI455X (gfx1250) — matrix operand order repaired and checked by reading, not compiled, not run on hardware
//
#include <hip/hip_runtime.h>

#define NN 50000
#define EE 800000
#define NRELC 4

typedef __attribute__((ext_vector_type(16))) __bf16 bf16x16;
typedef __attribute__((ext_vector_type(8)))  __bf16 bf16x8;
typedef __attribute__((ext_vector_type(8)))  float  f32x8;

// ---------- helpers ----------
__device__ __forceinline__ f32x8 wmma_bf16(bf16x16 a, bf16x16 b, f32x8 c) {
  // D = A(16x32 bf16) * B(32x16 bf16) + C(16x16 f32)
  return __builtin_amdgcn_wmma_f32_16x16x32_bf16(false, a, false, b, (short)0, c,
                                                 false, false);
}

// monotone map float -> uint so unsigned atomicMax == float max
__device__ __forceinline__ unsigned fmap(float f) {
  unsigned u = __float_as_uint(f);
  return (u & 0x80000000u) ? ~u : (u | 0x80000000u);
}
__device__ __forceinline__ float funmap(unsigned u) {
  return (u & 0x80000000u) ? __uint_as_float(u & 0x7FFFFFFFu)
                           : __uint_as_float(~u);
}

// A fragment (16x32 bf16): lane holds row M=L&15; halves map to
// K runs [hi*16, hi*16+8) and [hi*16+8, hi*16+16). Two aligned 16B loads.
__device__ __forceinline__ bf16x16 loadA(const __bf16* __restrict__ row, int hi) {
  bf16x8 lo = *(const bf16x8*)(row + hi * 16);
  bf16x8 hh = *(const bf16x8*)(row + hi * 16 + 8);
  return __builtin_shufflevector(lo, hh, 0, 1, 2, 3, 4, 5, 6, 7,
                                 8, 9, 10, 11, 12, 13, 14, 15);
}

// ---------- degree / cvt / norm ----------
__global__ __launch_bounds__(256) void k_degrees(const int* __restrict__ src,
                                                 const int* __restrict__ dst,
                                                 float* __restrict__ dout,
                                                 float* __restrict__ din) {
  int e = blockIdx.x * 256 + threadIdx.x;            // grid == EE/256 exactly
  atomicAdd(&dout[src[e]], 1.0f);
  atomicAdd(&din[dst[e]], 1.0f);
}

__global__ __launch_bounds__(256) void k_cvt(const float* __restrict__ in,
                                             __bf16* __restrict__ out) {
  long idx = (long)blockIdx.x * 256 + threadIdx.x;   // grid exact
  out[idx] = (__bf16)in[idx];
}

__global__ __launch_bounds__(256) void k_scale(const float* __restrict__ feat,
                                               const float* __restrict__ dout,
                                               __bf16* __restrict__ fsc) {
  int idx = blockIdx.x * 256 + threadIdx.x;          // grid == NN*32/256 exactly
  fsc[idx] = (__bf16)(feat[idx] * rsqrtf(fmaxf(dout[idx >> 5], 1.0f)));
}

// ---------- per-edge relation-typed GEMM: m[e] = xe @ Wr[eid[e]] ----------
// one wave per 16-edge tile; A rows masked per relation, C accumulated over r.
// Weights staged in LDS pre-swizzled into B-fragment register order so each
// lane fetches its 16 halves as one contiguous 32B load (2x ds_load_b128).
__global__ __launch_bounds__(256) void k_edge_gemm(
    const __bf16* __restrict__ fsc, const __bf16* __restrict__ efeat,
    const __bf16* __restrict__ nfeat, const int* __restrict__ src,
    const int* __restrict__ dst, const int* __restrict__ eid,
    const float* __restrict__ Wr, __bf16* __restrict__ mout) {
  __shared__ bf16x16 sWv[24 * 32];                   // 24 frags x 32 lanes x 32B
  {
    __bf16* sWh = (__bf16*)sWv;
    for (int idx = threadIdx.x; idx < NRELC * 96 * 32; idx += 256) {
      int r = idx / 3072, rem = idx - r * 3072;
      int k = rem >> 5, n = rem & 31;
      int c = k >> 5, kk = k & 31, hi = kk >> 4, i = kk & 15;
      int nb = n >> 4, nlo = n & 15;
      int lane = (hi << 4) | nlo;
      int frag = (r * 3 + c) * 2 + nb;
      sWh[((frag << 5) | lane) * 16 + i] = (__bf16)Wr[idx];
    }
  }
  __syncthreads();

  int tile = (blockIdx.x * 256 + threadIdx.x) >> 5;  // 50000 tiles exactly
  int lane = threadIdx.x & 31;
  int mrow = lane & 15, hi = lane >> 4, n0 = lane & 15;
  long ebase = (long)tile * 16;
  long e = ebase + mrow;
  int s = src[e], d = dst[e], re = eid[e];

  f32x8 acc0 = {0, 0, 0, 0, 0, 0, 0, 0};
  f32x8 acc1 = {0, 0, 0, 0, 0, 0, 0, 0};
  bf16x16 zv;
#pragma unroll
  for (int i = 0; i < 16; i++) zv[i] = (__bf16)0.0f;

#pragma unroll
  for (int c = 0; c < 3; c++) {                      // K chunks: src|edge|dst
    const __bf16* rowp = (c == 0) ? (fsc + (long)s * 32)
                        : (c == 1) ? (efeat + e * 32)
                                   : (nfeat + (long)d * 32);
    bf16x16 aC = loadA(rowp, hi);
#pragma unroll
    for (int r = 0; r < NRELC; r++) {
      bf16x16 aM = (re == r) ? aC : zv;              // row-mask (v_cndmask)
      bf16x16 b0 = sWv[(((r * 3 + c) * 2 + 0) << 5) | lane];
      bf16x16 b1 = sWv[(((r * 3 + c) * 2 + 1) << 5) | lane];
      acc0 = wmma_bf16(aM, b0, acc0);
      acc1 = wmma_bf16(aM, b1, acc1);
    }
  }
#pragma unroll
  for (int vi = 0; vi < 8; vi++) {
    long row = ebase + hi * 8 + vi;
    mout[row * 32 + n0]      = (__bf16)acc0[vi];
    mout[row * 32 + n0 + 16] = (__bf16)acc1[vi];
  }
}

// ---------- attention score + segment max ----------
__global__ __launch_bounds__(256) void k_attn(const float* __restrict__ nfeat,
                                              const int* __restrict__ dst,
                                              const __bf16* __restrict__ mbuf,
                                              const float* __restrict__ aw,
                                              float* __restrict__ zb,
                                              unsigned* __restrict__ zmax) {
  int e = blockIdx.x * 256 + threadIdx.x;
  int d = dst[e];
  float s = 0.0f;
#pragma unroll
  for (int j = 0; j < 32; j++) s += nfeat[(long)d * 32 + j] * aw[j];
#pragma unroll
  for (int j = 0; j < 32; j++) s += (float)mbuf[(long)e * 32 + j] * aw[32 + j];
  float z = (s > 0.0f) ? s : 0.01f * s;              // leaky_relu(0.01)
  zb[e] = z;
  atomicMax(&zmax[d], fmap(z));
}

// ---------- exp + segment sum ----------
__global__ __launch_bounds__(256) void k_expsum(const float* __restrict__ zb,
                                                const int* __restrict__ dst,
                                                const unsigned* __restrict__ zmax,
                                                float* __restrict__ ezb,
                                                float* __restrict__ sume) {
  int e = blockIdx.x * 256 + threadIdx.x;
  int d = dst[e];
  float v = __expf(zb[e] - funmap(zmax[d]));
  ezb[e] = v;
  atomicAdd(&sume[d], v);
}

// ---------- weighted scatter aggregation ----------
__global__ __launch_bounds__(256) void k_agg(const float* __restrict__ ezb,
                                             const float* __restrict__ sume,
                                             const int* __restrict__ dst,
                                             const __bf16* __restrict__ mbuf,
                                             float* __restrict__ hagg) {
  int idx = blockIdx.x * 256 + threadIdx.x;          // grid == EE*32/256 exactly
  int e = idx >> 5;
  int d = dst[e];
  float a = ezb[e] / sume[d];
  atomicAdd(&hagg[d * 32 + (idx & 31)], a * (float)mbuf[idx]);
}

// ---------- self-loop GEMM + degree norm + bias (1 wave / 16 nodes) ----------
__global__ __launch_bounds__(32) void k_selfloop(const __bf16* __restrict__ nfeat,
                                                 const float* __restrict__ lw,
                                                 const float* __restrict__ hagg,
                                                 const float* __restrict__ degin,
                                                 const float* __restrict__ hbias,
                                                 __bf16* __restrict__ hfin) {
  __shared__ bf16x16 sBv[2 * 32];
  int lane = threadIdx.x;
  for (int idx = lane; idx < 32 * 32; idx += 32) {   // swizzled weight stage
    int k = idx >> 5, n = idx & 31;
    int hi2 = k >> 4, i = k & 15, nb = n >> 4, nlo = n & 15;
    ((__bf16*)sBv)[((nb << 5) | ((hi2 << 4) | nlo)) * 16 + i] = (__bf16)lw[idx];
  }
  __syncthreads();
  int mrow = lane & 15, hi = lane >> 4, n0 = lane & 15;
  int base = blockIdx.x * 16;                        // NN/16 blocks exactly
  bf16x16 aF = loadA(nfeat + (long)(base + mrow) * 32, hi);
  bf16x16 b0 = sBv[lane], b1 = sBv[32 + lane];
  f32x8 acc0 = {0, 0, 0, 0, 0, 0, 0, 0};
  f32x8 acc1 = {0, 0, 0, 0, 0, 0, 0, 0};
  acc0 = wmma_bf16(aF, b0, acc0);
  acc1 = wmma_bf16(aF, b1, acc1);
  float bias0 = hbias[n0], bias1 = hbias[n0 + 16];
#pragma unroll
  for (int vi = 0; vi < 8; vi++) {
    int node = base + hi * 8 + vi;
    float sc = rsqrtf(fmaxf(degin[node], 1.0f));
    hfin[(long)node * 32 + n0] =
        (__bf16)((acc0[vi] + hagg[(long)node * 32 + n0]) * sc + bias0);
    hfin[(long)node * 32 + n0 + 16] =
        (__bf16)((acc1[vi] + hagg[(long)node * 32 + n0 + 16]) * sc + bias1);
  }
}

// ---------- AMRM: lv=relu(h@W_l+b_l); softmax over levels; out=relu(sum) ----------
__global__ __launch_bounds__(32) void k_amrm(const __bf16* __restrict__ hfin,
                                             const float* __restrict__ W,
                                             const float* __restrict__ bvec,
                                             const float* __restrict__ avec,
                                             float* __restrict__ outp) {
  __shared__ bf16x16 sBv[6 * 32];
  __shared__ float slv[16 * 3 * 32];                 // [node][level][out]
  int lane = threadIdx.x;
  for (int idx = lane; idx < 3 * 32 * 32; idx += 32) {  // swizzled weight stage
    int l = idx >> 10, rem = idx & 1023;
    int k = rem >> 5, n = rem & 31;
    int hi2 = k >> 4, i = k & 15, nb = n >> 4, nlo = n & 15;
    int frag = l * 2 + nb;
    ((__bf16*)sBv)[((frag << 5) | ((hi2 << 4) | nlo)) * 16 + i] = (__bf16)W[idx];
  }
  __syncthreads();
  int mrow = lane & 15, hi = lane >> 4, n0 = lane & 15;
  int base = blockIdx.x * 16;
  bf16x16 aF = loadA(hfin + (long)(base + mrow) * 32, hi);
#pragma unroll
  for (int l = 0; l < 3; l++) {
    bf16x16 b0 = sBv[(l * 2 + 0) * 32 + lane];
    bf16x16 b1 = sBv[(l * 2 + 1) * 32 + lane];
    f32x8 acc0 = {0, 0, 0, 0, 0, 0, 0, 0};
    f32x8 acc1 = {0, 0, 0, 0, 0, 0, 0, 0};
    acc0 = wmma_bf16(aF, b0, acc0);
    acc1 = wmma_bf16(aF, b1, acc1);
    float bb0 = bvec[l * 32 + n0], bb1 = bvec[l * 32 + n0 + 16];
#pragma unroll
    for (int vi = 0; vi < 8; vi++) {
      int m = vi + hi * 8;
      float v0 = acc0[vi] + bb0; v0 = v0 > 0.0f ? v0 : 0.0f;
      float v1 = acc1[vi] + bb1; v1 = v1 > 0.0f ? v1 : 0.0f;
      slv[(m * 3 + l) * 32 + n0]      = v0;
      slv[(m * 3 + l) * 32 + n0 + 16] = v1;
    }
  }
  __syncthreads();                                   // single-wave barrier
  if (lane < 16) {
    int node = base + lane;
    float s0 = 0, s1 = 0, s2 = 0;
    for (int o = 0; o < 32; o++) {
      float av = avec[o];
      s0 += slv[(lane * 3 + 0) * 32 + o] * av;
      s1 += slv[(lane * 3 + 1) * 32 + o] * av;
      s2 += slv[(lane * 3 + 2) * 32 + o] * av;
    }
    float mx = fmaxf(s0, fmaxf(s1, s2));
    float e0 = __expf(s0 - mx), e1 = __expf(s1 - mx), e2 = __expf(s2 - mx);
    float inv = 1.0f / (e0 + e1 + e2);
    for (int o = 0; o < 32; o++) {
      float v = (e0 * slv[(lane * 3 + 0) * 32 + o] +
                 e1 * slv[(lane * 3 + 1) * 32 + o] +
                 e2 * slv[(lane * 3 + 2) * 32 + o]) * inv;
      outp[(long)node * 32 + o] = v > 0.0f ? v : 0.0f;   // outer relu
    }
  }
}

extern "C" void kernel_launch(void* const* d_in, const int* in_sizes, int n_in,
                              void* d_out, int out_size, void* d_ws, size_t ws_size,
                              hipStream_t stream) {
  const float* x      = (const float*)d_in[0];
  const float* nfeat  = (const float*)d_in[1];
  const float* efeat  = (const float*)d_in[2];
  const int*   src    = (const int*)d_in[3];
  const int*   dst    = (const int*)d_in[4];
  const int*   eid    = (const int*)d_in[5];
  const float* Wr[2]   = {(const float*)d_in[6],  (const float*)d_in[13]};
  const float* attn[2] = {(const float*)d_in[7],  (const float*)d_in[14]};
  const float* lw[2]   = {(const float*)d_in[8],  (const float*)d_in[15]};
  const float* hb[2]   = {(const float*)d_in[9],  (const float*)d_in[16]};
  const float* aW[2]   = {(const float*)d_in[10], (const float*)d_in[17]};
  const float* ab[2]   = {(const float*)d_in[11], (const float*)d_in[18]};
  const float* aa[2]   = {(const float*)d_in[12], (const float*)d_in[19]};

  // workspace layout: 116*NN + 34*EE float-equivalents ~= 132 MB
  float* ws       = (float*)d_ws;
  float* deg_out  = ws;                         // NN
  float* deg_in   = ws + NN;                    // NN
  unsigned* zmax  = (unsigned*)(ws + 2 * NN);   // NN
  float* sume     = ws + 3 * NN;                // NN
  float* hagg     = ws + 4 * NN;                // 32*NN
  float* h1       = hagg + 32 * NN;             // 32*NN
  float* zb       = h1 + 32 * NN;               // EE
  float* ezb      = zb + EE;                    // EE
  __bf16* fsc_bf   = (__bf16*)(ezb + EE);       // NN*32 bf16
  __bf16* nfeat_bf = fsc_bf + (long)NN * 32;    // NN*32 bf16
  __bf16* hfin_bf  = nfeat_bf + (long)NN * 32;  // NN*32 bf16
  __bf16* efeat_bf = hfin_bf + (long)NN * 32;   // EE*32 bf16
  __bf16* mbuf_bf  = efeat_bf + (long)EE * 32;  // EE*32 bf16

  hipMemsetAsync(deg_out, 0, 2 * NN * sizeof(float), stream);
  k_degrees<<<EE / 256, 256, 0, stream>>>(src, dst, deg_out, deg_in);
  k_cvt<<<(NN * 32) / 256, 256, 0, stream>>>(nfeat, nfeat_bf);
  k_cvt<<<(EE * 32) / 256, 256, 0, stream>>>(efeat, efeat_bf);

  for (int L = 0; L < 2; L++) {
    const float* feat = (L == 0) ? x : h1;
    float* outp = (L == 0) ? h1 : (float*)d_out;
    hipMemsetAsync(zmax, 0, 2 * NN * sizeof(float), stream);   // zmax + sume
    hipMemsetAsync(hagg, 0, 32 * NN * sizeof(float), stream);
    k_scale<<<(NN * 32) / 256, 256, 0, stream>>>(feat, deg_out, fsc_bf);
    k_edge_gemm<<<EE / 16 / 8, 256, 0, stream>>>(fsc_bf, efeat_bf, nfeat_bf,
                                                 src, dst, eid, Wr[L], mbuf_bf);
    k_attn<<<EE / 256, 256, 0, stream>>>(nfeat, dst, mbuf_bf, attn[L], zb, zmax);
    k_expsum<<<EE / 256, 256, 0, stream>>>(zb, dst, zmax, ezb, sume);
    k_agg<<<(EE * 32) / 256, 256, 0, stream>>>(ezb, sume, dst, mbuf_bf, hagg);
    k_selfloop<<<NN / 16, 32, 0, stream>>>(nfeat_bf, lw[L], hagg, deg_in,
                                           hb[L], hfin_bf);
    k_amrm<<<NN / 16, 32, 0, stream>>>(hfin_bf, aW[L], ab[L], aa[L], outp);
  }
}
